// LazyDecoder_4612794876263
// MI455X (gfx1250) — matrix operand order repaired and checked by reading, not compiled, not run on hardware
//
#include <hip/hip_runtime.h>
#include <hip/hip_bf16.h>

typedef __attribute__((ext_vector_type(16))) _Float16 v16h;
typedef __attribute__((ext_vector_type(4)))  _Float16 v4h;
typedef __attribute__((ext_vector_type(8)))  float    v8f;

// ---------------- model dims ----------------
#define VV   32000
#define DD   768
#define HH   12
#define GKV  3
#define DFF  2048
#define EE   8
#define NLAY 4
#define NMOE 2
#define DHH  64
#define DCTX 256
#define BB   4
#define TT   512
#define TC   161            // 1 + 32 + 128 context tokens
#define TCP  164            // context tokens padded to a multiple of 4 (zero rows)
#define GDIM (GKV*DHH)      // 192
#define MT   (BB*TT)        // 2048

// ---------------- gemm flags ----------------
#define GF_TRANSB    1
#define GF_ACCUM     2
#define GF_SILU      4
#define GF_GATHER_A  8
#define GF_COMPACT_A 16
#define GF_COMPACT_C 32
#define GF_SCATTER_C 64
#define GF_BIAS      128
#define GF_SCALE     256

#define GF_MOE1 (GF_GATHER_A | GF_COMPACT_C | GF_SILU | GF_BIAS)
#define GF_MOE2 (GF_COMPACT_A | GF_SCATTER_C | GF_BIAS | GF_SCALE | GF_ACCUM)

struct GemmArgs {
  const float* A; const float* B; float* C;
  const float* bias; const float* scale;
  const int* rowIdx; const int* zoff; const int* zcnt;
  int M, N, K, lda, ldb, ldc;
  long long sAb, sAh, sBb, sBh, sCb, sCh;
  int Hdiv, bhDiv;
  long long biasStride;
  int flags;
};

// f32-in / f16-WMMA / f32-out GEMM, software-pipelined with double-buffered LDS.
// Block = 256 threads = 8 waves; block tile 128(M) x 64(N); K step 32.
// Waves: 4(M) x 2(N); each wave owns a 32x32 C tile: 2 A-frags x 2 B-frags
// -> 4 v_wmma per K step from 4 fragment reads (2 ds_load_b128 per WMMA).
// Requires K % 4 == 0 and N % 4 == 0 (the host pads context K/V to 164 rows),
// so every staging access is a single predicated 16B load + ds_store_b64 (4 b16 stores for [K,N] B).
// Fragment-order LDS layout (operand reads are ds_load_b128 pairs):
//   A: lane = hf*16+m; elem e -> K = hf*16+e
//     inverse: kk -> hf=(kk>>4)&1, e=kk&15
//   B (mirrors sparse-B table): lane = hf*16+n; elem e -> K = hf*16+e
//     inverse: kk -> hf=(kk>>4)&1, e=kk&15
//   C/D: lane holds col n = lane&15; VGPR vi -> row = hf*8 + vi
template <int FLAGS>
__global__ __launch_bounds__(256) void wmma_gemm_kernel(GemmArgs g) {
  __shared__ alignas(32) _Float16 fragA[2][8][32][16];  // [buf][m-tile][lane][elem]
  __shared__ alignas(32) _Float16 fragB[2][4][32][16];  // [buf][n-tile][lane][elem]

  const int z = blockIdx.z;
  const int b = z / g.Hdiv, h = z % g.Hdiv;
  const float* Ap = g.A + (long long)b * g.sAb + (long long)h * g.sAh;
  const float* Bp = g.B + (long long)b * g.sBb + (long long)(h / g.bhDiv) * g.sBh;
  float*       Cp = g.C + (long long)b * g.sCb + (long long)h * g.sCh;

  const int zo   = (FLAGS & (GF_GATHER_A | GF_COMPACT_A | GF_COMPACT_C | GF_SCATTER_C))
                       ? g.zoff[z] : 0;
  const int meff = g.zcnt ? g.zcnt[z] : g.M;
  const int m0 = blockIdx.x * 128, n0 = blockIdx.y * 64;
  if (m0 >= meff) return;  // grouped-MoE early exit (uniform per block)

  const int tid  = threadIdx.x;
  const int lane = tid & 31, wid = tid >> 5;
  const int wm = wid & 3, wn = wid >> 2;
  const int mr = lane & 15;

  v8f acc[2][2] = {{{}, {}}, {{}, {}}};  // [m-frag][n-frag]

  auto stage = [&](int k0, int sbuf) {
    // ---- A tile (128 rows x 32 k), fragment-order writes ----
#pragma unroll
    for (int it = 0; it < 4; ++it) {
      int idx  = tid + it * 256;       // 1024 items
      int r    = idx >> 3;             // 0..127
      int kk   = (idx & 7) * 4;        // 0,4,...,28
      int gk   = k0 + kk;
      int grow = m0 + r;
      float4 v = {0.f, 0.f, 0.f, 0.f};
      if (grow < meff && gk < g.K) {   // K%4==0 -> whole float4 in range
        int arow = grow;
        if (FLAGS & GF_GATHER_A)       arow = g.rowIdx[zo + grow];
        else if (FLAGS & GF_COMPACT_A) arow = zo + grow;
        v = *(const float4*)(Ap + (long long)arow * g.lda + gk);
      }
      int hf = (kk >> 4) & 1;
      int e  = kk & 15;
      v4h w;
      w[0] = (_Float16)v.x; w[1] = (_Float16)v.y;
      w[2] = (_Float16)v.z; w[3] = (_Float16)v.w;
      *(v4h*)&fragA[sbuf][r >> 4][hf * 16 + (r & 15)][e] = w;  // ds_store_b64
    }
    // ---- B tile (32 k x 64 n), fragment-order writes ----
    if (FLAGS & GF_TRANSB) {
      // B stored [N,K]: 16B loads along K, ds_store_b64
#pragma unroll
      for (int it = 0; it < 2; ++it) {
        int idx = tid + it * 256;
        int n   = idx >> 3;
        int kk  = (idx & 7) * 4;
        int gk  = k0 + kk;
        int gn  = n0 + n;
        float4 v = {0.f, 0.f, 0.f, 0.f};
        if (gn < g.N && gk < g.K)
          v = *(const float4*)(Bp + (long long)gn * g.ldb + gk);
        int hf = (kk >> 4) & 1;
        int e  = kk & 15;
        v4h w;
        w[0] = (_Float16)v.x; w[1] = (_Float16)v.y;
        w[2] = (_Float16)v.z; w[3] = (_Float16)v.w;
        *(v4h*)&fragB[sbuf][n >> 4][hf * 16 + (n & 15)][e] = w;
      }
    } else {
      // B stored [K,N]: 16B loads along N (N%4==0), lane-scattered b16 stores
#pragma unroll
      for (int it = 0; it < 2; ++it) {
        int idx = tid + it * 256;
        int kk  = idx >> 4;            // 0..31
        int nq  = (idx & 15) * 4;      // 0..60
        int gk  = k0 + kk;
        int gn  = n0 + nq;
        float4 v = {0.f, 0.f, 0.f, 0.f};
        if (gk < g.K && gn < g.N)
          v = *(const float4*)(Bp + (long long)gk * g.ldb + gn);
        int hf = (kk >> 4) & 1;
        int e  = kk & 15;
        fragB[sbuf][nq >> 4][hf * 16 + ((nq + 0) & 15)][e] = (_Float16)v.x;
        fragB[sbuf][nq >> 4][hf * 16 + ((nq + 1) & 15)][e] = (_Float16)v.y;
        fragB[sbuf][nq >> 4][hf * 16 + ((nq + 2) & 15)][e] = (_Float16)v.z;
        fragB[sbuf][nq >> 4][hf * 16 + ((nq + 3) & 15)][e] = (_Float16)v.w;
      }
    }
  };

  const int ktiles = (g.K + 31) >> 5;
  stage(0, 0);
  int cur = 0;
  for (int kt = 0; kt < ktiles; ++kt) {
    if (kt + 2 < ktiles)  // gfx1250 global_prefetch_b8, two tiles ahead
      __builtin_prefetch(Ap + (long long)(m0 + (tid >> 1)) * g.lda + ((kt + 2) << 5),
                         0, 1);
    __syncthreads();
    if (kt + 1 < ktiles) stage((kt + 1) << 5, cur ^ 1);

    v16h af0 = *(const v16h*)&fragA[cur][wm * 2 + 0][lane][0];
    v16h af1 = *(const v16h*)&fragA[cur][wm * 2 + 1][lane][0];
    v16h bf0 = *(const v16h*)&fragB[cur][wn * 2 + 0][lane][0];
    v16h bf1 = *(const v16h*)&fragB[cur][wn * 2 + 1][lane][0];
    acc[0][0] = __builtin_amdgcn_wmma_f32_16x16x32_f16(false, af0, false, bf0,
                                                       (short)0, acc[0][0], false, false);
    acc[0][1] = __builtin_amdgcn_wmma_f32_16x16x32_f16(false, af0, false, bf1,
                                                       (short)0, acc[0][1], false, false);
    acc[1][0] = __builtin_amdgcn_wmma_f32_16x16x32_f16(false, af1, false, bf0,
                                                       (short)0, acc[1][0], false, false);
    acc[1][1] = __builtin_amdgcn_wmma_f32_16x16x32_f16(false, af1, false, bf1,
                                                       (short)0, acc[1][1], false, false);
    cur ^= 1;
  }

  // ---- epilogue ----
  const float* biasp =
      (FLAGS & GF_BIAS) ? (g.bias + (long long)z * g.biasStride) : nullptr;
  const int hf = lane >> 4;
#pragma unroll
  for (int s = 0; s < 2; ++s) {
    int col = n0 + (wn * 2 + s) * 16 + mr;
    if (col >= g.N) continue;
    float bv = (FLAGS & GF_BIAS) ? biasp[col] : 0.f;
#pragma unroll
    for (int a = 0; a < 2; ++a) {
#pragma unroll
      for (int vi = 0; vi < 8; ++vi) {
        int row = m0 + (wm * 2 + a) * 16 + hf * 8 + vi;
        if (row >= meff) continue;
        float v = acc[a][s][vi] + bv;
        if (FLAGS & GF_SILU) v = v / (1.f + expf(-v));
        int crow;
        if (FLAGS & GF_COMPACT_C)      crow = zo + row;
        else if (FLAGS & GF_SCATTER_C) crow = g.rowIdx[zo + row];
        else                           crow = row;
        if (FLAGS & GF_SCALE) v *= g.scale[crow];
        float* cp = Cp + (long long)crow * g.ldc + col;
        if (FLAGS & GF_ACCUM) *cp += v; else *cp = v;
      }
    }
  }
}

// ---------------- small kernels ----------------
__global__ void embed_kernel(const int* __restrict__ ids,
                             const float* __restrict__ emb,
                             float* __restrict__ x, int D) {
  long long row = blockIdx.x;
  long long id  = ids[row];
  for (int i = threadIdx.x; i < D; i += blockDim.x)
    x[row * D + i] = emb[id * D + i];
}

__global__ __launch_bounds__(256) void rmsnorm_kernel(const float* __restrict__ X,
                                                      const float* __restrict__ W,
                                                      float* __restrict__ Y, int D) {
  long long row = blockIdx.x;
  const float* x = X + row * D;
  __shared__ float red[256];
  float s = 0.f;
  for (int i = threadIdx.x; i < D; i += 256) { float v = x[i]; s += v * v; }
  red[threadIdx.x] = s; __syncthreads();
  for (int o = 128; o > 0; o >>= 1) {
    if (threadIdx.x < o) red[threadIdx.x] += red[threadIdx.x + o];
    __syncthreads();
  }
  float inv = rsqrtf(red[0] / (float)D + 1e-6f);
  for (int i = threadIdx.x; i < D; i += 256)
    Y[row * D + i] = x[i] * inv * W[i];
}

// softmax over cols [0,N); writes zeros into pad cols [N,Npad)
__global__ __launch_bounds__(256) void softmax_kernel(float* __restrict__ S, int T,
                                                      int N, int Npad, int ld,
                                                      float scale, int causal) {
  long long row = blockIdx.x;
  int t = (int)(row % T);
  float* s = S + row * ld;
  __shared__ float red[256];
  float mx = -3.0e38f;
  for (int j = threadIdx.x; j < N; j += 256) {
    float v = s[j] * scale + ((causal && j > t) ? -1e9f : 0.f);
    mx = fmaxf(mx, v);
  }
  red[threadIdx.x] = mx; __syncthreads();
  for (int o = 128; o > 0; o >>= 1) {
    if (threadIdx.x < o) red[threadIdx.x] = fmaxf(red[threadIdx.x], red[threadIdx.x + o]);
    __syncthreads();
  }
  mx = red[0]; __syncthreads();
  float sum = 0.f;
  for (int j = threadIdx.x; j < N; j += 256) {
    float v = s[j] * scale + ((causal && j > t) ? -1e9f : 0.f);
    sum += expf(v - mx);
  }
  red[threadIdx.x] = sum; __syncthreads();
  for (int o = 128; o > 0; o >>= 1) {
    if (threadIdx.x < o) red[threadIdx.x] += red[threadIdx.x + o];
    __syncthreads();
  }
  float inv = 1.f / red[0]; __syncthreads();
  for (int j = threadIdx.x; j < Npad; j += 256) {
    float out = 0.f;
    if (j < N) {
      float v = s[j] * scale + ((causal && j > t) ? -1e9f : 0.f);
      out = expf(v - mx) * inv;
    }
    s[j] = out;
  }
}

__global__ void zero_float_kernel(float* p, long long n) {
  long long i = (long long)blockIdx.x * blockDim.x + threadIdx.x;
  if (i < n) p[i] = 0.f;
}

__global__ void zero_int_kernel(int* p, int n) {
  int i = blockIdx.x * blockDim.x + threadIdx.x;
  if (i < n) p[i] = 0;
}

__global__ void moe_route_kernel(const float* __restrict__ logits, int M,
                                 float* __restrict__ sc, int* __restrict__ eidx,
                                 int* __restrict__ pos, int* __restrict__ cnt) {
  int t = blockIdx.x * blockDim.x + threadIdx.x;
  if (t >= M) return;
  const float* g = logits + (long long)t * EE;
  float mx = g[0]; int be = 0;
  for (int e = 1; e < EE; ++e) if (g[e] > mx) { mx = g[e]; be = e; }
  float sum = 0.f;
  for (int e = 0; e < EE; ++e) sum += expf(g[e] - mx);
  sc[t]   = 1.f / sum;  // softmax max prob
  eidx[t] = be;
  pos[t]  = atomicAdd(&cnt[be], 1);
}

__global__ void moe_scan_kernel(const int* __restrict__ cnt, int* __restrict__ offs) {
  if (threadIdx.x == 0) {
    int a = 0;
    for (int e = 0; e < EE; ++e) { offs[e] = a; a += cnt[e]; }
  }
}

__global__ void moe_fill_kernel(const int* __restrict__ eidx,
                                const int* __restrict__ pos,
                                const int* __restrict__ offs,
                                int* __restrict__ perm, int M) {
  int t = blockIdx.x * blockDim.x + threadIdx.x;
  if (t < M) perm[offs[eidx[t]] + pos[t]] = t;
}

// ---------------- host ----------------
static void launch_gemm(const GemmArgs& ga, int Z, hipStream_t stream) {
  dim3 grid((ga.M + 127) / 128, (ga.N + 63) / 64, Z);
  dim3 blk(256);
  switch (ga.flags) {
    case 0:
      wmma_gemm_kernel<0><<<grid, blk, 0, stream>>>(ga); break;
    case GF_TRANSB:
      wmma_gemm_kernel<GF_TRANSB><<<grid, blk, 0, stream>>>(ga); break;
    case GF_ACCUM:
      wmma_gemm_kernel<GF_ACCUM><<<grid, blk, 0, stream>>>(ga); break;
    case GF_SILU:
      wmma_gemm_kernel<GF_SILU><<<grid, blk, 0, stream>>>(ga); break;
    case GF_MOE1:
      wmma_gemm_kernel<GF_MOE1><<<grid, blk, 0, stream>>>(ga); break;
    case GF_MOE2:
      wmma_gemm_kernel<GF_MOE2><<<grid, blk, 0, stream>>>(ga); break;
    default: break;
  }
}

extern "C" void kernel_launch(void* const* d_in, const int* in_sizes, int n_in,
                              void* d_out, int out_size, void* d_ws, size_t ws_size,
                              hipStream_t stream) {
  (void)in_sizes; (void)n_in; (void)out_size; (void)ws_size;
  const int*   ids      = (const int*)  d_in[0];
  const float* usr      = (const float*)d_in[1];
  const float* sht      = (const float*)d_in[2];
  const float* lng      = (const float*)d_in[3];
  const float* tok_emb  = (const float*)d_in[4];
  const float* ctx_proj = (const float*)d_in[5];
  const float* ctx_nk   = (const float*)d_in[6];
  const float* normq_w  = (const float*)d_in[7];
  const float* qproj_w  = (const float*)d_in[8];
  const float* cattn_w  = (const float*)d_in[9];
  const float* normqkv  = (const float*)d_in[10];
  const float* qkv_w    = (const float*)d_in[11];
  const float* sattn_w  = (const float*)d_in[12];
  const float* ffn_nw   = (const float*)d_in[13];
  const float* ffn_w1   = (const float*)d_in[14];
  const float* ffn_w2   = (const float*)d_in[15];
  const float* moe_nw   = (const float*)d_in[16];
  const float* gate_w   = (const float*)d_in[17];
  const float* moe_w1   = (const float*)d_in[18];
  const float* moe_b1   = (const float*)d_in[19];
  const float* moe_w2   = (const float*)d_in[20];
  const float* moe_b2   = (const float*)d_in[21];
  const float* out_nw   = (const float*)d_in[22];
  const float* out_pw   = (const float*)d_in[23];

  char* wp = (char*)d_ws;
  auto alloc = [&](size_t elems, size_t esz) -> void* {
    void* r = (void*)wp;
    wp += ((elems * esz + 255) & ~(size_t)255);
    return r;
  };
  float* x      = (float*)alloc((size_t)MT * DD, 4);
  float* xn     = (float*)alloc((size_t)MT * DD, 4);
  float* q      = (float*)alloc((size_t)MT * DD, 4);
  float* qkvb   = (float*)alloc((size_t)MT * 3 * DD, 4);
  float* attnb  = (float*)alloc((size_t)MT * DD, 4);
  float* ctxp   = (float*)alloc((size_t)BB * TCP * GDIM, 4);  // padded rows
  float* kc     = (float*)alloc((size_t)BB * TCP * GDIM, 4);  // padded rows
  float* scores = (float*)alloc((size_t)BB * HH * TT * TT, 4);
  float* hbuf   = (float*)alloc((size_t)MT * DFF, 4);
  float* gateb  = (float*)alloc((size_t)MT * EE, 4);
  float* scb    = (float*)alloc((size_t)MT, 4);
  int*   eidxb  = (int*)alloc((size_t)MT, 4);
  int*   posb   = (int*)alloc((size_t)MT, 4);
  int*   permb  = (int*)alloc((size_t)MT, 4);
  int*   cnt    = (int*)alloc(EE, 4);
  int*   offs   = (int*)alloc(EE, 4);

  auto G = [](const float* A, const float* B, float* C, int M_, int N_, int K_,
              int lda, int ldb, int ldc, int flags) {
    GemmArgs ga = {};
    ga.A = A; ga.B = B; ga.C = C;
    ga.M = M_; ga.N = N_; ga.K = K_;
    ga.lda = lda; ga.ldb = ldb; ga.ldc = ldc;
    ga.Hdiv = 1; ga.bhDiv = 1; ga.flags = flags;
    return ga;
  };
  auto rms = [&](const float* X, const float* W, float* Y, int rows, int D) {
    rmsnorm_kernel<<<dim3(rows), dim3(256), 0, stream>>>(X, W, Y, D);
  };

  // ---- context processor: [us;st;lt] @ ctx_proj -> ctxp[B,164,192] (rows 161..163 zero)
  {
    long long n = (long long)BB * TCP * GDIM;
    zero_float_kernel<<<dim3((unsigned)((n + 255) / 256)), dim3(256), 0, stream>>>(ctxp, n);
  }
  {
    GemmArgs ga = G(usr, ctx_proj, ctxp, 1, GDIM, DCTX, DCTX, GDIM, GDIM, 0);
    ga.sAb = DCTX; ga.sCb = (long long)TCP * GDIM; launch_gemm(ga, BB, stream);
  }
  {
    GemmArgs ga = G(sht, ctx_proj, ctxp + GDIM, 32, GDIM, DCTX, DCTX, GDIM, GDIM, 0);
    ga.sAb = 32 * DCTX; ga.sCb = (long long)TCP * GDIM; launch_gemm(ga, BB, stream);
  }
  {
    GemmArgs ga = G(lng, ctx_proj, ctxp + 33 * GDIM, 128, GDIM, DCTX, DCTX, GDIM, GDIM, 0);
    ga.sAb = 128 * DCTX; ga.sCb = (long long)TCP * GDIM; launch_gemm(ga, BB, stream);
  }
  rms(ctxp, ctx_nk, kc, BB * TCP, GDIM);  // zero rows stay zero

  // ---- token embedding gather
  embed_kernel<<<dim3(MT), dim3(256), 0, stream>>>(ids, tok_emb, x, DD);

  for (int l = 0; l < NLAY; ++l) {
    // ======== cross attention (GQA: head h reads kv group h/4) ========
    rms(x, normq_w + (size_t)l * DD, xn, MT, DD);
    launch_gemm(G(xn, qproj_w + (size_t)l * DD * DD, q, MT, DD, DD, DD, DD, DD, 0),
                1, stream);
    {
      GemmArgs ga = G(q, kc, scores, TT, TCP, DHH, DD, GDIM, TCP, GF_TRANSB);
      ga.Hdiv = HH; ga.bhDiv = HH / GKV;
      ga.sAb = (long long)TT * DD;    ga.sAh = DHH;
      ga.sBb = (long long)TCP * GDIM; ga.sBh = DHH;
      ga.sCb = (long long)HH * TT * TCP; ga.sCh = (long long)TT * TCP;
      launch_gemm(ga, BB * HH, stream);
    }
    softmax_kernel<<<dim3(BB * HH * TT), dim3(256), 0, stream>>>(
        scores, TT, TC, TCP, TCP, 0.125f, 0);
    {
      // K runs over padded 164 context rows: pad probs are 0, pad kc rows are 0
      GemmArgs ga = G(scores, kc, attnb, TT, DHH, TCP, TCP, GDIM, DD, 0);
      ga.Hdiv = HH; ga.bhDiv = HH / GKV;
      ga.sAb = (long long)HH * TT * TCP; ga.sAh = (long long)TT * TCP;
      ga.sBb = (long long)TCP * GDIM;    ga.sBh = DHH;
      ga.sCb = (long long)TT * DD;       ga.sCh = DHH;
      launch_gemm(ga, BB * HH, stream);
    }
    launch_gemm(G(attnb, cattn_w + (size_t)l * DD * DD, x, MT, DD, DD, DD, DD, DD,
                  GF_ACCUM), 1, stream);

    // ======== causal self attention ========
    rms(x, normqkv + (size_t)l * DD, xn, MT, DD);
    launch_gemm(G(xn, qkv_w + (size_t)l * DD * 3 * DD, qkvb, MT, 3 * DD, DD,
                  DD, 3 * DD, 3 * DD, 0), 1, stream);
    {
      GemmArgs ga = G(qkvb, qkvb + DD, scores, TT, TT, DHH, 3 * DD, 3 * DD, TT,
                      GF_TRANSB);
      ga.Hdiv = HH;
      ga.sAb = (long long)TT * 3 * DD; ga.sAh = DHH;
      ga.sBb = (long long)TT * 3 * DD; ga.sBh = DHH;
      ga.sCb = (long long)HH * TT * TT; ga.sCh = (long long)TT * TT;
      launch_gemm(ga, BB * HH, stream);
    }
    softmax_kernel<<<dim3(BB * HH * TT), dim3(256), 0, stream>>>(
        scores, TT, TT, TT, TT, 0.125f, 1);
    {
      GemmArgs ga = G(scores, qkvb + 2 * DD, attnb, TT, DHH, TT, TT, 3 * DD, DD, 0);
      ga.Hdiv = HH;
      ga.sAb = (long long)HH * TT * TT; ga.sAh = (long long)TT * TT;
      ga.sBb = (long long)TT * 3 * DD;  ga.sBh = DHH;
      ga.sCb = (long long)TT * DD;      ga.sCh = DHH;
      launch_gemm(ga, BB * HH, stream);
    }
    launch_gemm(G(attnb, sattn_w + (size_t)l * DD * DD, x, MT, DD, DD, DD, DD, DD,
                  GF_ACCUM), 1, stream);

    if (l < NLAY - NMOE) {
      // ======== dense FFN: x += silu(xn@w1) @ w2 ========
      int i = l;
      rms(x, ffn_nw + (size_t)i * DD, xn, MT, DD);
      launch_gemm(G(xn, ffn_w1 + (size_t)i * DD * DFF, hbuf, MT, DFF, DD,
                    DD, DFF, DFF, GF_SILU), 1, stream);
      launch_gemm(G(hbuf, ffn_w2 + (size_t)i * DFF * DD, x, MT, DD, DFF,
                    DFF, DD, DD, GF_ACCUM), 1, stream);
    } else {
      // ======== top-1 MoE as grouped gather-GEMM ========
      int i = l - (NLAY - NMOE);
      rms(x, moe_nw + (size_t)i * DD, xn, MT, DD);
      launch_gemm(G(xn, gate_w + (size_t)i * DD * EE, gateb, MT, EE, DD,
                    DD, EE, EE, 0), 1, stream);
      zero_int_kernel<<<dim3(1), dim3(32), 0, stream>>>(cnt, EE);
      moe_route_kernel<<<dim3((MT + 255) / 256), dim3(256), 0, stream>>>(
          gateb, MT, scb, eidxb, posb, cnt);
      moe_scan_kernel<<<dim3(1), dim3(1), 0, stream>>>(cnt, offs);
      moe_fill_kernel<<<dim3((MT + 255) / 256), dim3(256), 0, stream>>>(
          eidxb, posb, offs, permb, MT);
      {
        GemmArgs ga = G(xn, moe_w1 + (size_t)i * EE * DD * DFF, hbuf, MT, DFF, DD,
                        DD, DFF, DFF, GF_MOE1);
        ga.sBb = (long long)DD * DFF;
        ga.bias = moe_b1 + (size_t)i * EE * DFF; ga.biasStride = DFF;
        ga.rowIdx = permb; ga.zoff = offs; ga.zcnt = cnt;
        launch_gemm(ga, EE, stream);
      }
      {
        GemmArgs ga = G(hbuf, moe_w2 + (size_t)i * EE * DFF * DD, x, MT, DD, DFF,
                        DFF, DD, DD, GF_MOE2);
        ga.sBb = (long long)DFF * DD;
        ga.bias = moe_b2 + (size_t)i * EE * DD; ga.biasStride = DD;
        ga.rowIdx = permb; ga.zoff = offs; ga.zcnt = cnt; ga.scale = scb;
        launch_gemm(ga, EE, stream);
      }
    }
  }

  // ---- output head: rmsnorm then 2048x32000x768 WMMA GEMM into d_out
  rms(x, out_nw, xn, MT, DD);
  launch_gemm(G(xn, out_pw, (float*)d_out, MT, VV, DD, DD, VV, VV, 0), 1, stream);
}
